// PQLinear_89593017794939
// MI455X (gfx1250) — matrix operand order repaired and checked by reading, not compiled, not run on hardware
//
#include <hip/hip_runtime.h>
#include <stdint.h>

// ---------------------------------------------------------------------------
// PQ-quantized linear:  out[M,O] = x[M,I] @ wq[O,I]^T + bias
//   wq = codebook[argmin ||w/rs - c||^2] * rs      (per 8-wide group)
// M=4096 (B*S), I=2048, O=2048, K=256 codewords, D=8.
// Path: f32 quantize -> bf16 operand buffers in d_ws -> v_wmma_f32_16x16x32_bf16
// Staging: GLOBAL_LOAD_ASYNC_TO_LDS_B128 (ASYNCcnt) when available.
// ---------------------------------------------------------------------------

typedef __attribute__((ext_vector_type(16))) __bf16 v16bf;
typedef __attribute__((ext_vector_type(8)))  float  v8f;

#define AS1 __attribute__((address_space(1)))
#define AS3 __attribute__((address_space(3)))

#if defined(__has_builtin)
#if __has_builtin(__builtin_amdgcn_global_load_async_to_lds_b128)
#define HAVE_ASYNC_LDS 1
#endif
#endif
#ifndef HAVE_ASYNC_LDS
#define HAVE_ASYNC_LDS 0
#endif

// exact parameter type of the async-LDS builtins (per hipcc diagnostic)
typedef int async_v4i __attribute__((vector_size(4 * sizeof(int))));

static __device__ __forceinline__ unsigned int f2bf(float f) {
  unsigned int u = __float_as_uint(f);
  unsigned int r = u + 0x7FFFu + ((u >> 16) & 1u);   // round-to-nearest-even
  return r >> 16;
}

// copy 32B global -> LDS (per lane); async DMA on CDNA5, sync fallback else
static __device__ __forceinline__ void cp32_g2l(const unsigned short* g,
                                                unsigned short* l) {
#if HAVE_ASYNC_LDS
  AS1 async_v4i* gp = (AS1 async_v4i*)g;   // addrspacecast + type cast
  AS3 async_v4i* lp = (AS3 async_v4i*)l;
  __builtin_amdgcn_global_load_async_to_lds_b128(gp, lp, 0, 0);
  __builtin_amdgcn_global_load_async_to_lds_b128(gp, lp, 16, 0);
#else
  const uint4* pg = (const uint4*)g;
  uint4* pl = (uint4*)l;
  uint4 t0 = pg[0], t1 = pg[1];
  pl[0] = t0; pl[1] = t1;
#endif
}

static __device__ __forceinline__ void wait_async_all() {
#if HAVE_ASYNC_LDS
#if __has_builtin(__builtin_amdgcn_s_wait_asynccnt)
  __builtin_amdgcn_s_wait_asynccnt(0);
#else
  asm volatile("s_wait_asynccnt 0x0" ::: "memory");
#endif
#endif
}

// ---------------- kernel 1: x (f32) -> xb (bf16), 8 elems/thread ------------
__global__ __launch_bounds__(256) void cvt_x_kernel(
    const float* __restrict__ in, unsigned short* __restrict__ out, int n8) {
  int t = blockIdx.x * 256 + threadIdx.x;
  if (t >= n8) return;
  const float4* p = (const float4*)(in + (size_t)t * 8);
  float4 a = p[0], b = p[1];
  uint4 o;
  o.x = f2bf(a.x) | (f2bf(a.y) << 16);
  o.y = f2bf(a.z) | (f2bf(a.w) << 16);
  o.z = f2bf(b.x) | (f2bf(b.y) << 16);
  o.w = f2bf(b.z) | (f2bf(b.w) << 16);
  ((uint4*)out)[t] = o;
}

// ---------------- kernel 2: PQ quantize weight -> wqb (bf16) ----------------
__global__ __launch_bounds__(256) void pq_quant_kernel(
    const float* __restrict__ weight, const float* __restrict__ codebook,
    const float* __restrict__ row_scale, unsigned short* __restrict__ wqb) {
  __shared__ float cb[256 * 8];
  __shared__ float cn[256];
  const int t = threadIdx.x;

  float4 c0 = ((const float4*)codebook)[t * 2];
  float4 c1 = ((const float4*)codebook)[t * 2 + 1];
  ((float4*)cb)[t * 2]     = c0;
  ((float4*)cb)[t * 2 + 1] = c1;
  cn[t] = 0.5f * (c0.x * c0.x + c0.y * c0.y + c0.z * c0.z + c0.w * c0.w +
                  c1.x * c1.x + c1.y * c1.y + c1.z * c1.z + c1.w * c1.w);
  __syncthreads();

  const size_t g = (size_t)blockIdx.x * 256 + t;    // group id, [0, O*I/8)
  const int o = (int)(g >> 8);                      // I/D = 256 groups per row
  const float scale = row_scale[o];
  const float inv = 1.0f / scale;

  const float4* wp = (const float4*)(weight + g * 8);
  float4 w0 = wp[0], w1 = wp[1];
  float gv0 = w0.x * inv, gv1 = w0.y * inv, gv2 = w0.z * inv, gv3 = w0.w * inv;
  float gv4 = w1.x * inv, gv5 = w1.y * inv, gv6 = w1.z * inv, gv7 = w1.w * inv;

  float best = 3.4e38f;
  int bi = 0;
  for (int k = 0; k < 256; ++k) {                   // argmin 0.5||c||^2 - g.c
    const float* c = cb + k * 8;
    float s = cn[k];
    s -= gv0 * c[0]; s -= gv1 * c[1]; s -= gv2 * c[2]; s -= gv3 * c[3];
    s -= gv4 * c[4]; s -= gv5 * c[5]; s -= gv6 * c[6]; s -= gv7 * c[7];
    if (s < best) { best = s; bi = k; }
  }

  const float* c = cb + bi * 8;
  uint4 o16;
  o16.x = f2bf(c[0] * scale) | (f2bf(c[1] * scale) << 16);
  o16.y = f2bf(c[2] * scale) | (f2bf(c[3] * scale) << 16);
  o16.z = f2bf(c[4] * scale) | (f2bf(c[5] * scale) << 16);
  o16.w = f2bf(c[6] * scale) | (f2bf(c[7] * scale) << 16);
  ((uint4*)wqb)[g] = o16;
}

// ---------------- kernel 3: bf16 WMMA GEMM + bias ---------------------------
#define BM 128
#define BN 128
#define BK 32
#define LDSTR 40   // 32 + 8 bf16 pad -> 80B row stride, conflict-free b128 rows

static __device__ __forceinline__ v16bf ldfrag(const unsigned short* p0,
                                               const unsigned short* p1) {
  union { uint4 u[2]; v16bf v; } r;
  r.u[0] = *(const uint4*)p0;
  r.u[1] = *(const uint4*)p1;
  return r.v;
}

__global__ __launch_bounds__(256) void gemm_bf16_kernel(
    const unsigned short* __restrict__ A,   // xb  [M,K] bf16 row-major
    const unsigned short* __restrict__ B,   // wqb [N,K] bf16 row-major
    const float* __restrict__ bias,         // [N]
    float* __restrict__ C,                  // [M,N] f32
    int M, int N, int K) {
  __shared__ unsigned short sA[2][BM * LDSTR];
  __shared__ unsigned short sB[2][BN * LDSTR];

  const int tid   = threadIdx.x;
  const int lane  = tid & 31;
  const int wave  = tid >> 5;      // 8 waves
  const int waveM = wave >> 2;     // 0..1  -> 64 rows each
  const int waveN = wave & 3;      // 0..3  -> 32 cols each
  const int m0 = blockIdx.y * BM;
  const int n0 = blockIdx.x * BN;

  // staging: each thread moves 32B per matrix: row = tid>>1, half = tid&1
  const int lr = tid >> 1;
  const int lh = tid & 1;
  const unsigned short* gA = A + (size_t)(m0 + lr) * K + lh * 16;
  const unsigned short* gB = B + (size_t)(n0 + lr) * K + lh * 16;
  const int sOff = lr * LDSTR + lh * 16;

  const int frow  = lane & 15;
  const int khalf = lane >> 4;

  const v8f vzero = {0.f, 0.f, 0.f, 0.f, 0.f, 0.f, 0.f, 0.f};
  v8f acc[4][2];
#pragma unroll
  for (int i = 0; i < 4; ++i)
#pragma unroll
    for (int j = 0; j < 2; ++j) acc[i][j] = vzero;

  // prologue: DMA tile 0 into buffer 0
  cp32_g2l(gA, &sA[0][sOff]);
  cp32_g2l(gB, &sB[0][sOff]);
  wait_async_all();
  __syncthreads();

  const int NK = K / BK;   // 64
  int buf = 0;
  for (int kk = 0; kk < NK; ++kk) {
    if (kk + 1 < NK) {     // DMA next tile into the other buffer
      const int nb = buf ^ 1;
      cp32_g2l(gA + (size_t)(kk + 1) * BK, &sA[nb][sOff]);
      cp32_g2l(gB + (size_t)(kk + 1) * BK, &sB[nb][sOff]);
    }

    const unsigned short* aBase = &sA[buf][0];
    const unsigned short* bBase = &sB[buf][0];

    // A frag (16x32 bf16): per-lane 16B chunks at k = khalf*8 and +16
    v16bf afrag[4];
#pragma unroll
    for (int i = 0; i < 4; ++i) {
      const unsigned short* p =
          aBase + (waveM * 64 + i * 16 + frow) * LDSTR + khalf * 8;
      afrag[i] = ldfrag(p, p + 16);
    }
    // B frag (32x16 bf16): per-lane 16B chunks at k = khalf*8 and +16
    v16bf bfrag[2];
#pragma unroll
    for (int j = 0; j < 2; ++j) {
      const unsigned short* p =
          bBase + (waveN * 32 + j * 16 + frow) * LDSTR + khalf * 8;
      bfrag[j] = ldfrag(p, p + 16);
    }

#pragma unroll
    for (int i = 0; i < 4; ++i)
#pragma unroll
      for (int j = 0; j < 2; ++j)
        acc[i][j] = __builtin_amdgcn_wmma_f32_16x16x32_bf16(
            false, afrag[i], false, bfrag[j], (short)0, acc[i][j],
            false, false);

    wait_async_all();      // our async writes into buf^1 have landed
    __syncthreads();       // everyone's reads of buf + writes of buf^1 done
    buf ^= 1;
  }

  // epilogue: D layout (16x16 f32): VGPR v -> M = v + 8*(lane>>4), N = lane&15
  const int mbase = m0 + waveM * 64 + 8 * khalf;
  const int nbase = n0 + waveN * 32 + frow;
#pragma unroll
  for (int i = 0; i < 4; ++i) {
#pragma unroll
    for (int j = 0; j < 2; ++j) {
      const int nb = nbase + j * 16;
      const float bv = bias[nb];
      const int idx = (mbase + i * 16) * N + nb;    // fits in int32 (<= 2^23)
#pragma unroll
      for (int v = 0; v < 8; ++v) C[idx + v * N] = acc[i][j][v] + bv;
    }
  }
}

// ---------------------------------------------------------------------------
extern "C" void kernel_launch(void* const* d_in, const int* in_sizes, int n_in,
                              void* d_out, int out_size, void* d_ws,
                              size_t ws_size, hipStream_t stream) {
  const float* x         = (const float*)d_in[0];   // [B,S,I] = [M,I]
  const float* weight    = (const float*)d_in[1];   // [O,I]
  const float* codebook  = (const float*)d_in[2];   // [256,8]
  const float* row_scale = (const float*)d_in[3];   // [O,1]
  const float* bias      = (const float*)d_in[4];   // [O]

  const int I = 2048, O = 2048;
  const int M = in_sizes[0] / I;                    // 4096

  unsigned short* xb  = (unsigned short*)d_ws;      // [M,I] bf16 (16 MB)
  unsigned short* wqb = xb + (size_t)M * I;         // [O,I] bf16 (8 MB)

  // 1) x -> bf16
  const int n8 = (M * I) / 8;
  cvt_x_kernel<<<n8 / 256, 256, 0, stream>>>(x, xb, n8);

  // 2) PQ quantize + dequant weight -> bf16
  const int ngroups = (O * I) / 8;                  // 524288
  pq_quant_kernel<<<ngroups / 256, 256, 0, stream>>>(weight, codebook,
                                                     row_scale, wqb);

  // 3) bf16 WMMA GEMM + bias -> f32 out
  dim3 grid(O / BN, M / BM);                        // (16, 32)
  gemm_bf16_kernel<<<grid, 256, 0, stream>>>(xb, wqb, bias, (float*)d_out,
                                             M, O, I);
}
